// cate_mixture_model_27951647162952
// MI455X (gfx1250) — matrix operand order repaired and checked by reading, not compiled, not run on hardware
//
#include <hip/hip_runtime.h>
#include <math.h>

// CDNA5 / gfx1250 categorical mixture log-likelihood via one-hot WMMA GEMM.
// K=16 mixtures, D=128 dims, C=64 categories, N=65536 samples.

typedef __attribute__((ext_vector_type(16))) _Float16 v16h;
typedef __attribute__((ext_vector_type(8)))  float    v8f;

#define KMIX 16
#define DDIM 128
#define CCAT 64

union V16U { v16h v; unsigned int u[8]; int4 q[2]; };

__device__ __forceinline__ unsigned int pack_f16x2(float a, float b) {
    _Float16 ha = (_Float16)a, hb = (_Float16)b;
    unsigned short ua, ub;
    __builtin_memcpy(&ua, &ha, 2);
    __builtin_memcpy(&ub, &hb, 2);
    return (unsigned int)ua | ((unsigned int)ub << 16);
}

// ---------------------------------------------------------------------------
// Prep: per-dim d, compute logp = log_softmax(cate[k,d,:]) for all k,
// shift by s[d] = max_{k,c} logp, and store f16 residuals pre-swizzled in the
// wave32 WMMA B-matrix layout, contiguous k order (same as A):
//   B tile t (cats 32t..32t+31): lane L holds column n = L&15,
//   rows K = rb + 2j, 2j+1 in dword j (rb = 0 for lanes 0-15, 16 for 16-31).
// Bbuf dword index: (d*2 + t)*256 + L*8 + j  -> per-lane 32B contiguous.
// ---------------------------------------------------------------------------
__global__ void cmm_prep(const float* __restrict__ cate,
                         unsigned int* __restrict__ Bbuf,
                         float* __restrict__ s_arr)
{
    const int d    = blockIdx.x;       // 0..127
    const int lane = threadIdx.x;      // 0..31 (one wave per block)

    float va[KMIX], vb[KMIX], denom[KMIX];
    float smax = -__builtin_inff();

    #pragma unroll
    for (int k = 0; k < KMIX; ++k) {
        const float* row = cate + (size_t)(k * DDIM + d) * CCAT;
        float a = row[lane];
        float b = row[lane + 32];
        va[k] = a; vb[k] = b;
        float mx = fmaxf(a, b);
        #pragma unroll
        for (int m = 16; m >= 1; m >>= 1) mx = fmaxf(mx, __shfl_xor(mx, m, 32));
        float e = __expf(a - mx) + __expf(b - mx);
        #pragma unroll
        for (int m = 16; m >= 1; m >>= 1) e += __shfl_xor(e, m, 32);
        float lse = __logf(e);          // >= 0
        denom[k] = mx + lse;            // logp = v - denom
        smax = fmaxf(smax, -lse);       // max_c logp[k,d,:] = -lse
    }

    __shared__ float res[KMIX][CCAT + 2];   // +2 pad vs bank conflicts
    #pragma unroll
    for (int k = 0; k < KMIX; ++k) {
        res[k][lane]      = va[k] - denom[k] - smax;
        res[k][lane + 32] = vb[k] - denom[k] - smax;
    }
    __syncthreads();

    const int mix = lane & 15;
    const int rb  = (lane >> 4) << 4;   // 0 or 16
    #pragma unroll
    for (int t = 0; t < 2; ++t) {
        #pragma unroll
        for (int j = 0; j < 8; ++j) {
            int c0 = t * 32 + rb + 2 * j;
            Bbuf[(d * 2 + t) * 256 + lane * 8 + j] =
                pack_f16x2(res[mix][c0], res[mix][c0 + 1]);
        }
    }
    if (lane == 0) s_arr[d] = smax;
}

// ---------------------------------------------------------------------------
// logwS[k] = log_softmax(mixture_weight)[k] + sum_d s[d]
// ---------------------------------------------------------------------------
__global__ void cmm_logw(const float* __restrict__ mw,
                         const float* __restrict__ s_arr,
                         float* __restrict__ logwS)
{
    const int lane = threadIdx.x;
    float S = 0.f;
    #pragma unroll
    for (int i = 0; i < 4; ++i) S += s_arr[i * 32 + lane];
    #pragma unroll
    for (int m = 16; m >= 1; m >>= 1) S += __shfl_xor(S, m, 32);

    float w = (lane < KMIX) ? mw[lane] : -__builtin_inff();
    float mx = w;
    #pragma unroll
    for (int m = 16; m >= 1; m >>= 1) mx = fmaxf(mx, __shfl_xor(mx, m, 32));
    float e = __expf(w - mx);                 // exp(-inf)=0 for idle lanes
    #pragma unroll
    for (int m = 16; m >= 1; m >>= 1) e += __shfl_xor(e, m, 32);
    if (lane < KMIX) logwS[lane] = (w - mx - __logf(e)) + S;
}

// ---------------------------------------------------------------------------
// Main: one wave per 16-sample tile; per dim d, two v_wmma_f32_16x16x32_f16
// with one-hot A (from LDS lookup table) and pre-swizzled B (from L2).
// Two independent accumulators (cats 0-31 / 32-63) break the WMMA RAW chain.
// ---------------------------------------------------------------------------
__global__ void __launch_bounds__(128)
cmm_main(const int* __restrict__ x,
         const unsigned int* __restrict__ Bbuf,
         const float* __restrict__ logwS,
         float* __restrict__ out)
{
    // x tile per wave: 16 rows x 128 ints, row stride 132 (bank-conflict pad)
    __shared__ __align__(16) int ldsx[4][16 * 132];
    // one-hot dword LUT: T[t][j] = onehot dword for u = (t-64) - off_j,
    // row stride 12 dwords (b128-aligned, spreads start banks)
    __shared__ __align__(16) unsigned int T[128 * 12];

    const int lane = threadIdx.x & 31;
    const int w    = threadIdx.x >> 5;
    const int tile = blockIdx.x * 4 + w;
    const int s0   = tile * 16;
    int* myx = ldsx[w];

    // Build LUT (128 threads, one row each). off_j in contiguous k order (matches B):
    // dword j holds K = off_j, off_j+1 with off_j = 2j (lanes 16-31 add 16 via half).
    {
        int i = threadIdx.x;
        int t = i - 64;
        #pragma unroll
        for (int j = 0; j < 8; ++j) {
            int off = j << 1;
            unsigned int u = (unsigned int)(t - off);
            T[i * 12 + j] = (u < 2u) ? (0x3C00u << ((u & 1u) << 4)) : 0u;
        }
    }

    // Cooperative coalesced x-tile load into padded LDS.
    {
        const int4* xg = (const int4*)(x + (size_t)s0 * DDIM);
        #pragma unroll
        for (int row = 0; row < 16; ++row) {
            int4 v = xg[row * 32 + lane];
            *(int4*)&myx[row * 132 + lane * 4] = v;
        }
    }
    __syncthreads();

    const int m    = lane & 15;
    const int half = (lane >> 4) << 4;      // K-offset fold for lanes 16-31
    v8f acc0 = {0.f, 0.f, 0.f, 0.f, 0.f, 0.f, 0.f, 0.f};
    v8f acc1 = {0.f, 0.f, 0.f, 0.f, 0.f, 0.f, 0.f, 0.f};

    #pragma unroll 1
    for (int dg = 0; dg < 32; ++dg) {
        int4 xq = *(const int4*)&myx[m * 132 + dg * 4];
        __builtin_prefetch(Bbuf + (size_t)(((dg * 4 + 16) & 127)) * 512 + lane * 8, 0, 0);
        #pragma unroll
        for (int qi = 0; qi < 4; ++qi) {
            int d  = dg * 4 + qi;
            int xv = (qi == 0) ? xq.x : (qi == 1) ? xq.y : (qi == 2) ? xq.z : xq.w;
            int t0 = xv - half;             // tile0 index; tile1 = t0 - 32

            const unsigned int* r0 = &T[(t0 + 64) * 12];   // cats 0..31
            const unsigned int* r1 = &T[(t0 + 32) * 12];   // cats 32..63
            V16U a0, a1;
            a0.q[0] = *(const int4*)(r0);
            a0.q[1] = *(const int4*)(r0 + 4);
            a1.q[0] = *(const int4*)(r1);
            a1.q[1] = *(const int4*)(r1 + 4);

            const unsigned int* bp = Bbuf + (size_t)d * 512 + lane * 8;
            v16h b0 = *(const v16h*)(bp);
            v16h b1 = *(const v16h*)(bp + 256);

            acc0 = __builtin_amdgcn_wmma_f32_16x16x32_f16(
                       false, a0.v, false, b0, (short)0, acc0, false, false);
            acc1 = __builtin_amdgcn_wmma_f32_16x16x32_f16(
                       false, a1.v, false, b1, (short)0, acc1, false, false);
        }
    }

    // Epilogue: D[M=sample, N=mixture]; lane holds mixture = lane&15,
    // samples r (+8 for lanes>=16) across the 8 acc regs.
    float lw = logwS[m];
    float res[8];
    #pragma unroll
    for (int r = 0; r < 8; ++r) {
        float v = acc0[r] + acc1[r] + lw;
        float mx = v;
        #pragma unroll
        for (int msk = 8; msk >= 1; msk >>= 1)
            mx = fmaxf(mx, __shfl_xor(mx, msk, 32));
        float e = __expf(v - mx);
        #pragma unroll
        for (int msk = 8; msk >= 1; msk >>= 1)
            e += __shfl_xor(e, msk, 32);
        res[r] = mx + __logf(e);
    }
    if ((lane & 15) == 0) {
        int base = s0 + ((lane >> 4) << 3);
        #pragma unroll
        for (int r = 0; r < 8; ++r) out[base + r] = res[r];
    }
}

// ---------------------------------------------------------------------------
extern "C" void kernel_launch(void* const* d_in, const int* in_sizes, int n_in,
                              void* d_out, int out_size, void* d_ws, size_t ws_size,
                              hipStream_t stream)
{
    (void)n_in; (void)out_size; (void)ws_size;
    const int*   x    = (const int*)d_in[0];          // [N, 128] int32
    const float* mw   = (const float*)d_in[1];        // [16] f32
    const float* cate = (const float*)d_in[2];        // [16, 128, 64] f32
    float* out = (float*)d_out;                       // [N] f32

    // Workspace: Bbuf 256KB (65536 dwords) + s_arr[128] + logwS[16]
    unsigned int* Bbuf  = (unsigned int*)d_ws;
    float*        s_arr = (float*)d_ws + 65536;
    float*        logwS = s_arr + 128;

    const int N = in_sizes[0] / DDIM;                 // 65536

    cmm_prep<<<DDIM, 32, 0, stream>>>(cate, Bbuf, s_arr);
    cmm_logw<<<1, 32, 0, stream>>>(mw, s_arr, logwS);
    cmm_main<<<N / 64, 128, 0, stream>>>(x, Bbuf, logwS, out);
}
